// FavorSelfAttention_60782377173093
// MI455X (gfx1250) — matrix operand order repaired and checked by reading, not compiled, not run on hardware
//
#include <hip/hip_runtime.h>
#include <stddef.h>

// FAVOR+ (Performer) causal linear attention for MI455X (gfx1250, wave32).
// Chunked formulation: all matmuls via v_wmma_f32_16x16x32_bf16, state kept f32.
// Chunk tiles staged LDS-side via the Tensor Data Mover (tensor_load_to_lds).

#define BB 4
#define LL 2048
#define HH 16
#define DD 64
#define MM 128
#define CC 64                 // chunk length along L
#define NCHUNK (LL / CC)
#define FEPS 1e-6f
#define SCALE 0.35355339059327373f   // 64^-0.25
#define RATIO 0.08838834764831845f   // 128^-0.5

typedef __bf16 bf16;
typedef __attribute__((ext_vector_type(16))) __bf16 v16bf;
typedef __attribute__((ext_vector_type(8)))  float  v8f;
typedef unsigned int __attribute__((ext_vector_type(4))) v4u;
typedef int  __attribute__((ext_vector_type(4))) v4i;
typedef int  __attribute__((ext_vector_type(8))) v8i;

// LDS leading dims (padded)
#define LDX 72     // 64-wide bf16 tiles
#define LDP 72     // projection rows
#define LDF 136    // 128-wide bf16 feature tiles
#define LDS_S 68   // f32 state rows

#define WMMA_BF16(a, b, c) \
  __builtin_amdgcn_wmma_f32_16x16x32_bf16(false, (a), false, (b), (short)0, (c), false, false)

// ---- Tensor Data Mover: load a 64x64 f32 tile (row stride H*D elems) to LDS ----
// D# bitfields per CDNA5 ISA.
__device__ __forceinline__ void tdm_load_tile64x64_f32(unsigned lds_byte_addr, const void* gptr) {
  unsigned long long ga = (unsigned long long)(size_t)gptr;
  v4u g0;
  g0[0] = 1u;                                           // count=1, user descriptor
  g0[1] = lds_byte_addr;                                // LDS dest byte address
  g0[2] = (unsigned)(ga & 0xffffffffu);                 // global_addr[31:0]
  g0[3] = (unsigned)((ga >> 32) & 0x01ffffffu) | (2u << 30);  // global_addr[56:32], type=2
  v8i g1;
  g1[0] = (int)(2u << 16);   // workgroup_mask=0, data_size=2 (4 bytes)
  g1[1] = (int)(64u << 16);  // tensor_dim0 = 64 (elems per row)
  g1[2] = (int)(64u << 16);  // tensor_dim1 = 64 (rows)
  g1[3] = (int)(64u << 16);  // tile_dim0 = 64
  g1[4] = 64;                // tile_dim1 = 64, tile_dim2 = 0
  g1[5] = (int)(HH * DD);    // tensor_dim0_stride = 1024 elems
  g1[6] = 0;                 // stride hi, dim1_stride lo (unused for 2D)
  g1[7] = 0;
  v4i z4 = {0, 0, 0, 0};
#if defined(__clang_major__) && (__clang_major__ >= 23)
  v8i z8 = {0, 0, 0, 0, 0, 0, 0, 0};
  __builtin_amdgcn_tensor_load_to_lds(g0, g1, z4, z4, z8, 0);
#else
  __builtin_amdgcn_tensor_load_to_lds(g0, g1, z4, z4, 0);
#endif
}

// ---- fragment loaders (per CDNA5 ISA VGPR layouts, wave32) ----

// A-operand 16x32: lane holds row (row0 + lane&15); K map per ISA table.
__device__ __forceinline__ v16bf frag_rowK(const bf16* base, int ld, int row0, int k0, int lane) {
  const bf16* p = base + (row0 + (lane & 15)) * ld + k0;
  int h = (lane >> 4) & 1;
  v16bf f;
#pragma unroll
  for (int e = 0; e < 16; ++e) {
    int pi = e >> 1;
    int K = ((pi >> 2) << 4) + (h << 3) + ((pi & 3) << 1) + (e & 1);
    f[e] = p[K];
  }
  return f;
}

// A-operand where logical A[m,k] = src[k][m] (transposed storage).
__device__ __forceinline__ v16bf fragA_colmajor(const bf16* base, int ld, int row0, int k0, int lane) {
  int r = row0 + (lane & 15);
  int h = (lane >> 4) & 1;
  v16bf f;
#pragma unroll
  for (int e = 0; e < 16; ++e) {
    int pi = e >> 1;
    int K = ((pi >> 2) << 4) + (h << 3) + ((pi & 3) << 1) + (e & 1);
    f[e] = base[(k0 + K) * ld + r];
  }
  return f;
}

// B-operand 32x16, B stored row-major [k][n]: K map per ISA table (same as A).
__device__ __forceinline__ v16bf fragB_rowmajor(const bf16* base, int ld, int col0, int k0, int lane) {
  int n = col0 + (lane & 15);
  int h = (lane >> 4) & 1;
  v16bf f;
#pragma unroll
  for (int e = 0; e < 16; ++e) {
    int pi = e >> 1;
    int K = ((pi >> 2) << 4) + (h << 3) + ((pi & 3) << 1) + (e & 1);
    f[e] = base[(k0 + K) * ld + n];
  }
  return f;
}

// B-operand where logical B[k,n] = src[n][k] (we hold the transpose row-major).
__device__ __forceinline__ v16bf fragB_fromT(const bf16* base, int ld, int col0, int k0, int lane) {
  const bf16* p = base + (col0 + (lane & 15)) * ld + k0;
  int h = (lane >> 4) & 1;
  v16bf f;
#pragma unroll
  for (int e = 0; e < 16; ++e) {
    int pi = e >> 1;
    int K = ((pi >> 2) << 4) + (h << 3) + ((pi & 3) << 1) + (e & 1);
    f[e] = p[K];
  }
  return f;
}

// ---------------- Pass 1: global max of k data_dash per (b,h) ----------------

__global__ void __launch_bounds__(256)
favor_kmax_kernel(const float* __restrict__ kin, const float* __restrict__ proj,
                  float* __restrict__ kmax_out) {
  __shared__ float stageK[CC * DD];
  __shared__ bf16  Pl[MM * LDP];
  __shared__ bf16  Xk[CC * LDX];
  __shared__ float red[8];

  int tid = threadIdx.x, lane = tid & 31;
  int wave = __builtin_amdgcn_readfirstlane(tid >> 5);
  int bh = blockIdx.x, b = bh / HH, h = bh % HH;

  for (int i = tid; i < MM * DD; i += 256) {
    int m = i >> 6, d = i & 63;
    Pl[m * LDP + d] = (bf16)proj[m * DD + d];
  }

  int r = tid >> 2, c0 = (tid & 3) << 4;
  float mx = -3.0e38f;
  for (int chunk = 0; chunk < NCHUNK; ++chunk) {
    __syncthreads();  // previous consumers of stage/Xk done
    if (wave == 0) {
      const float* base = kin + (size_t)((b * LL + chunk * CC) * HH + h) * DD;
      tdm_load_tile64x64_f32((unsigned)(size_t)(void*)stageK, base);
      __builtin_amdgcn_s_wait_tensorcnt(0);
    }
    __syncthreads();
#pragma unroll
    for (int j = 0; j < 16; ++j)
      Xk[r * LDX + c0 + j] = (bf16)(SCALE * stageK[r * DD + c0 + j]);
    __syncthreads();

    for (int t = wave; t < 32; t += 8) {   // 4x8 tiles of [64,128] dash
      int ti = t >> 3, tj = t & 7;
      v8f c = {0.f, 0.f, 0.f, 0.f, 0.f, 0.f, 0.f, 0.f};
      for (int kk = 0; kk < DD; kk += 32) {
        v16bf a  = frag_rowK(Xk, LDX, ti * 16, kk, lane);
        v16bf bb = fragB_fromT(Pl, LDP, tj * 16, kk, lane);
        c = WMMA_BF16(a, bb, c);
      }
#pragma unroll
      for (int vv = 0; vv < 8; ++vv) mx = fmaxf(mx, c[vv]);
    }
  }
#pragma unroll
  for (int off = 16; off; off >>= 1) mx = fmaxf(mx, __shfl_xor(mx, off, 32));
  if (lane == 0) red[wave] = mx;
  __syncthreads();
  if (tid == 0) {
    float m2 = red[0];
#pragma unroll
    for (int i = 1; i < 8; ++i) m2 = fmaxf(m2, red[i]);
    kmax_out[bh] = m2;
  }
}

// ---------------- Pass 2: chunked causal linear attention ----------------

__global__ void __launch_bounds__(256)
favor_main_kernel(const float* __restrict__ qin, const float* __restrict__ kin,
                  const float* __restrict__ vin, const float* __restrict__ proj,
                  const float* __restrict__ kmax_in, float* __restrict__ out) {
  __shared__ float stageQ[CC * DD];     // TDM staging (raw f32)
  __shared__ float stageK[CC * DD];
  __shared__ float stageV[CC * DD];
  __shared__ bf16  Pl[MM * LDP];        // projection (bf16)
  __shared__ bf16  Xq[CC * LDX];        // scaled q chunk
  __shared__ bf16  Xk[CC * LDX];        // scaled k chunk
  __shared__ bf16  Vb[CC * LDX];        // v chunk
  __shared__ float QDf[CC * MM];        // raw q dash (f32, for row max)
  __shared__ bf16  QDb[CC * LDF];       // q' features
  __shared__ bf16  KDb[CC * LDF];       // k' features
  __shared__ bf16  Ab[CC * LDX];        // masked intra-chunk attention
  __shared__ float S[MM * LDS_S];       // running state K'^T V (f32)
  __shared__ bf16  Sb[MM * LDX];        // bf16 mirror of S for WMMA B-operand
  __shared__ float diagq[CC], diagk[CC], rmax[CC], den[CC], dsum[MM];

  int tid = threadIdx.x, lane = tid & 31;
  int wave = __builtin_amdgcn_readfirstlane(tid >> 5);
  int bh = blockIdx.x, b = bh / HH, h = bh % HH;
  float kmax = kmax_in[bh];

  for (int i = tid; i < MM * DD; i += 256) {
    int m = i >> 6, d = i & 63;
    Pl[m * LDP + d] = (bf16)proj[m * DD + d];
  }
  for (int i = tid; i < MM * LDS_S; i += 256) S[i] = 0.f;
  for (int i = tid; i < MM * LDX; i += 256) Sb[i] = (bf16)0.f;
  if (tid < MM) dsum[tid] = 0.f;

  int r = tid >> 2, c0 = (tid & 3) << 4;  // 4 threads per chunk row

  for (int chunk = 0; chunk < NCHUNK; ++chunk) {
    __syncthreads();  // previous chunk's consumers of staged LDS done

    // --- step 0: TDM async tile loads (q,k,v) into LDS staging ---
    if (wave == 0) {
      size_t tb = (size_t)((b * LL + chunk * CC) * HH + h) * DD;
      tdm_load_tile64x64_f32((unsigned)(size_t)(void*)stageQ, qin + tb);
      tdm_load_tile64x64_f32((unsigned)(size_t)(void*)stageK, kin + tb);
      tdm_load_tile64x64_f32((unsigned)(size_t)(void*)stageV, vin + tb);
      __builtin_amdgcn_s_wait_tensorcnt(0);
    }
    __syncthreads();

    // --- step 1: scale/convert to bf16, compute diag terms ---
    float sq = 0.f, sk = 0.f;
#pragma unroll
    for (int j = 0; j < 16; ++j) {
      float xq = SCALE * stageQ[r * DD + c0 + j];
      float xk = SCALE * stageK[r * DD + c0 + j];
      Xq[r * LDX + c0 + j] = (bf16)xq;
      Xk[r * LDX + c0 + j] = (bf16)xk;
      Vb[r * LDX + c0 + j] = (bf16)stageV[r * DD + c0 + j];
      sq += xq * xq;
      sk += xk * xk;
    }
    sq += __shfl_xor(sq, 1, 32); sq += __shfl_xor(sq, 2, 32);
    sk += __shfl_xor(sk, 1, 32); sk += __shfl_xor(sk, 2, 32);
    if ((tid & 3) == 0) { diagq[r] = 0.5f * sq; diagk[r] = 0.5f * sk; }
    if (chunk + 1 < NCHUNK) {  // GL2 prefetch of next chunk (global_prefetch_b8)
      size_t nb = (size_t)((b * LL + (chunk + 1) * CC + r) * HH + h) * DD + c0;
      __builtin_prefetch(&qin[nb], 0, 0);
      __builtin_prefetch(&kin[nb], 0, 0);
      __builtin_prefetch(&vin[nb], 0, 0);
    }
    __syncthreads();

    // --- step 2: dash matmuls (q -> QDf raw; k -> exp features KDb) ---
    for (int job = wave; job < 64; job += 8) {
      int which = job >> 5, t = job & 31;
      int ti = t >> 3, tj = t & 7;
      const bf16* X = which ? Xk : Xq;
      v8f c = {0.f, 0.f, 0.f, 0.f, 0.f, 0.f, 0.f, 0.f};
      for (int kk = 0; kk < DD; kk += 32) {
        v16bf a  = frag_rowK(X, LDX, ti * 16, kk, lane);
        v16bf bb = fragB_fromT(Pl, LDP, tj * 16, kk, lane);
        c = WMMA_BF16(a, bb, c);
      }
      int n = tj * 16 + (lane & 15), hh = (lane >> 4) & 1;
      if (which == 0) {
#pragma unroll
        for (int vv = 0; vv < 8; ++vv) QDf[(ti * 16 + hh * 8 + vv) * MM + n] = c[vv];
      } else {
#pragma unroll
        for (int vv = 0; vv < 8; ++vv) {
          int row = ti * 16 + hh * 8 + vv;
          float kp = RATIO * (__expf(c[vv] - diagk[row] - kmax) + FEPS);
          KDb[row * LDF + n] = (bf16)kp;
        }
      }
    }
    __syncthreads();

    // --- step 3a: per-row max of raw q dash ---
    {
      int m0 = (tid & 3) << 5;
      float mxr = -3.0e38f;
#pragma unroll
      for (int j = 0; j < 32; ++j) mxr = fmaxf(mxr, QDf[r * MM + m0 + j]);
      mxr = fmaxf(mxr, __shfl_xor(mxr, 1, 32));
      mxr = fmaxf(mxr, __shfl_xor(mxr, 2, 32));
      if ((tid & 3) == 0) rmax[r] = mxr;
    }
    __syncthreads();

    // --- step 3b: q' features + inter-chunk denominator  den = q' . dsum_prev ---
    {
      int m0 = (tid & 3) << 5;
      float sub = diagq[r] + rmax[r];
      float acc = 0.f;
#pragma unroll
      for (int j = 0; j < 32; ++j) {
        int m = m0 + j;
        float qp = RATIO * (__expf(QDf[r * MM + m] - sub) + FEPS);
        QDb[r * LDF + m] = (bf16)qp;
        acc += qp * dsum[m];
      }
      acc += __shfl_xor(acc, 1, 32);
      acc += __shfl_xor(acc, 2, 32);
      if ((tid & 3) == 0) den[r] = acc;
    }
    __syncthreads();

    // --- step 4: intra-chunk A = Q' K'^T, causal mask, rowsum -> den ---
    for (int t = wave; t < 16; t += 8) {
      int ti = t >> 2, tj = t & 3;
      v8f c = {0.f, 0.f, 0.f, 0.f, 0.f, 0.f, 0.f, 0.f};
      if (tj <= ti) {  // strictly-upper tiles are all-masked
        for (int kk = 0; kk < MM; kk += 32) {
          v16bf a  = frag_rowK(QDb, LDF, ti * 16, kk, lane);
          v16bf bb = fragB_fromT(KDb, LDF, tj * 16, kk, lane);
          c = WMMA_BF16(a, bb, c);
        }
      }
      int n = lane & 15, hh = (lane >> 4) & 1;
      float rs[8];
#pragma unroll
      for (int vv = 0; vv < 8; ++vv) {
        int i = ti * 16 + hh * 8 + vv, jcol = tj * 16 + n;
        float val = (jcol <= i) ? c[vv] : 0.f;   // inclusive diagonal (ref updates s,dsum first)
        Ab[i * LDX + jcol] = (bf16)val;
        rs[vv] = val;
      }
#pragma unroll
      for (int vv = 0; vv < 8; ++vv) {
#pragma unroll
        for (int off = 1; off < 16; off <<= 1) rs[vv] += __shfl_xor(rs[vv], off, 32);
      }
      if (n == 0) {
#pragma unroll
        for (int vv = 0; vv < 8; ++vv) atomicAdd(&den[ti * 16 + hh * 8 + vv], rs[vv]);
      }
    }
    __syncthreads();

    // --- step 5: NUM = A.V + Q'.S_prev ; write out / den ---
    for (int t = wave; t < 16; t += 8) {
      int ti = t >> 2, tj = t & 3;
      v8f c = {0.f, 0.f, 0.f, 0.f, 0.f, 0.f, 0.f, 0.f};
      for (int kk = 0; kk < CC; kk += 32) {
        v16bf a  = frag_rowK(Ab, LDX, ti * 16, kk, lane);
        v16bf bb = fragB_rowmajor(Vb, LDX, tj * 16, kk, lane);
        c = WMMA_BF16(a, bb, c);
      }
      for (int kk = 0; kk < MM; kk += 32) {
        v16bf a  = frag_rowK(QDb, LDF, ti * 16, kk, lane);
        v16bf bb = fragB_rowmajor(Sb, LDX, tj * 16, kk, lane);
        c = WMMA_BF16(a, bb, c);
      }
      int n = lane & 15, hh = (lane >> 4) & 1;
#pragma unroll
      for (int vv = 0; vv < 8; ++vv) {
        int i = ti * 16 + hh * 8 + vv;
        int l = chunk * CC + i;
        out[(size_t)((b * LL + l) * HH + h) * DD + tj * 16 + n] = c[vv] / den[i];
      }
    }
    __syncthreads();

    // --- step 6: S += K'^T V (f32 accumulate); refresh Sb; dsum += colsum(K') ---
    for (int t = wave; t < 32; t += 8) {
      int ti = t >> 2, tj = t & 3;   // ti: M tiles (8), tj: D tiles (4)
      int n = lane & 15, hh = (lane >> 4) & 1;
      v8f c;
#pragma unroll
      for (int vv = 0; vv < 8; ++vv) c[vv] = S[(ti * 16 + hh * 8 + vv) * LDS_S + tj * 16 + n];
      for (int kk = 0; kk < CC; kk += 32) {
        v16bf a  = fragA_colmajor(KDb, LDF, ti * 16, kk, lane);  // (K'^T)[m,t] = KDb[t][m]
        v16bf bb = fragB_rowmajor(Vb, LDX, tj * 16, kk, lane);
        c = WMMA_BF16(a, bb, c);
      }
#pragma unroll
      for (int vv = 0; vv < 8; ++vv) {
        int mrow = ti * 16 + hh * 8 + vv;
        S[mrow * LDS_S + tj * 16 + n] = c[vv];
        Sb[mrow * LDX + tj * 16 + n] = (bf16)c[vv];
      }
    }
    if (tid < MM) {
      float acc = 0.f;
#pragma unroll 8
      for (int t2 = 0; t2 < CC; ++t2) acc += (float)KDb[t2 * LDF + tid];
      dsum[tid] += acc;
    }
  }
}

extern "C" void kernel_launch(void* const* d_in, const int* in_sizes, int n_in,
                              void* d_out, int out_size, void* d_ws, size_t ws_size,
                              hipStream_t stream) {
  const float* q    = (const float*)d_in[0];
  const float* k    = (const float*)d_in[1];
  const float* v    = (const float*)d_in[2];
  const float* proj = (const float*)d_in[3];
  float* out  = (float*)d_out;
  float* kmax = (float*)d_ws;  // 64 floats

  favor_kmax_kernel<<<BB * HH, 256, 0, stream>>>(k, proj, kmax);
  favor_main_kernel<<<BB * HH, 256, 0, stream>>>(q, k, v, proj, kmax, out);
}
